// Model_10608569221458
// MI455X (gfx1250) — matrix operand order repaired and checked by reading, not compiled, not run on hardware
//
#include <hip/hip_runtime.h>
#include <hip/hip_bf16.h>
#include <math.h>

// ---------------- types ----------------
typedef __attribute__((ext_vector_type(16))) __bf16 v16bf;
typedef __attribute__((ext_vector_type(8)))  float  v8f;
typedef unsigned int u32;

union Frag { v16bf v; u32 u[8]; };

// ---------------- model constants ----------------
#define BB   32
#define CH   1024
#define PATCH 196
#define SS   588          // sequence length 3*14*14
#define SP   608          // padded to multiple of 32
#define EE   512
#define VV   30000
#define TD   31           // decoder steps (T-1)
#define NH_  8
#define MM   (BB*SS)      // 18816 rows (multiple of 16)
#define QT   37           // ceil(588/16)

__device__ __forceinline__ float sigf(float x){ return 1.0f/(1.0f+__expf(-x)); }

// ---------------- elementwise / packing kernels ----------------
__global__ void cvt_f32_bf16(const float* __restrict__ s, __bf16* __restrict__ d, size_t n){
  size_t i = (size_t)blockIdx.x*blockDim.x + threadIdx.x;
  if (i < n) d[i] = (__bf16)s[i];
}
__global__ void copy_f32(const float* __restrict__ s, float* __restrict__ d, size_t n){
  size_t i = (size_t)blockIdx.x*blockDim.x + threadIdx.x;
  if (i < n) d[i] = s[i];
}
// Build A matrix for projection GEMM: rows = [i1; i2; i1-i2] per batch, transposed to (tokens, C)
__global__ void pack_img(const float* __restrict__ I1, const float* __restrict__ I2,
                         __bf16* __restrict__ A){
  size_t i = (size_t)blockIdx.x*blockDim.x + threadIdx.x; // over MM*CH
  if (i >= (size_t)MM*CH) return;
  int m = (int)(i >> 10);
  int c = (int)(i & 1023);
  int b = m / SS, p = m % SS;
  int seg = p / PATCH, pp = p % PATCH;
  size_t src = ((size_t)b*CH + c)*PATCH + pp;
  float v = (seg==0) ? I1[src] : (seg==1) ? I2[src] : (I1[src]-I2[src]);
  A[i] = (__bf16)v;
}
__global__ void gather_emb(const float* __restrict__ we, const int* __restrict__ des,
                           __bf16* __restrict__ xs){
  int row = blockIdx.x;  // t*BB + b, TD*BB rows
  int e = threadIdx.x;   // 512
  int t = row / BB, b = row % BB;
  int tok = des[b*(TD+1) + t];
  xs[(size_t)row*EE + e] = (__bf16)we[(size_t)tok*EE + e];
}
__global__ void mean_rows(const float* __restrict__ X, float* __restrict__ G){
  int b = blockIdx.x; int e = threadIdx.x; // 512 threads
  float s = 0.f;
  const float* xp = X + ((size_t)b*SS)*EE + e;
  for (int i=0;i<SS;i++) s += xp[(size_t)i*EE];
  G[b*EE+e] = s * (1.0f/SS);
}

// ---------------- LayerNorm (E=512, block=256, 2 elems/thread) ----------------
__global__ __launch_bounds__(256) void ln_kernel(const float* __restrict__ x,
    const float* __restrict__ ga, const float* __restrict__ be,
    float* __restrict__ outF, __bf16* __restrict__ outB)
{
  __shared__ float sh[8];
  int row = blockIdx.x;
  const float* xr = x + (size_t)row*EE;
  int t0 = threadIdx.x, t1 = threadIdx.x+256;
  float v0 = xr[t0], v1 = xr[t1];
  float s = v0+v1;
  for (int o=16;o;o>>=1) s += __shfl_xor(s,o,32);
  if ((threadIdx.x&31)==0) sh[threadIdx.x>>5] = s;
  __syncthreads();
  if (threadIdx.x==0){ float tot=0.f; for(int i=0;i<8;i++) tot+=sh[i]; sh[0]=tot; }
  __syncthreads();
  float mean = sh[0]*(1.0f/EE);
  __syncthreads();
  float d0=v0-mean, d1=v1-mean;
  s = d0*d0+d1*d1;
  for (int o=16;o;o>>=1) s += __shfl_xor(s,o,32);
  if ((threadIdx.x&31)==0) sh[threadIdx.x>>5] = s;
  __syncthreads();
  if (threadIdx.x==0){ float tot=0.f; for(int i=0;i<8;i++) tot+=sh[i]; sh[0]=tot; }
  __syncthreads();
  float var = sh[0]*(1.0f/(EE-1));
  float inv = 1.0f/(sqrtf(var)+1e-6f);
  float r0 = ga[t0]*d0*inv + be[t0];
  float r1 = ga[t1]*d1*inv + be[t1];
  if (outF){ outF[(size_t)row*EE+t0]=r0; outF[(size_t)row*EE+t1]=r1; }
  outB[(size_t)row*EE+t0]=(__bf16)r0;
  outB[(size_t)row*EE+t1]=(__bf16)r1;
}

// ---------------- generic WMMA GEMM: out = A(MxK) * W(NxK)^T ----------------
enum { EPI_F32=0, EPI_SIGPOS, EPI_RELU_BF16, EPI_RESID, EPI_QKV, EPI_VT, EPI_TANH };

template<int EPI>
__global__ __launch_bounds__(256) void gemm_bf16(
    const __bf16* __restrict__ A, int lda,
    const __bf16* __restrict__ W,
    const float* __restrict__ bias,
    int M, int N, int K,
    float* __restrict__ outF, int ldF,
    __bf16* __restrict__ outB, int ldB,
    const float* __restrict__ aux)
{
  int ntil = (N+63)>>6;
  int wid = blockIdx.x*8 + (threadIdx.x>>5);
  if (wid >= (M>>4)*ntil) return;
  int mt = wid / ntil, nt = wid % ntil;
  int lane = threadIdx.x&31, l15 = lane&15, half = lane>>4;
  const __bf16* arow = A + (size_t)(mt*16 + l15)*lda;

  v8f acc[4];
  #pragma unroll
  for (int c=0;c<4;c++){ v8f z = {0.f,0.f,0.f,0.f,0.f,0.f,0.f,0.f}; acc[c]=z; }

  for (int k0=0;k0<K;k0+=32){
    Frag a;
    #pragma unroll
    for (int j=0;j<8;j++){
      int kk = k0 + 2*(j&3) + 16*(j>>2) + 8*half;   // A-operand layout
      a.u[j] = *(const u32*)(arow + kk);
    }
    #pragma unroll
    for (int c=0;c<4;c++){
      int n = nt*64 + c*16 + l15;
      int nc = (n<N)?n:(N-1);
      const __bf16* wrow = W + (size_t)nc*K;
      Frag b;
      #pragma unroll
      for (int j=0;j<8;j++){
        int kk = k0 + 2*(j&3) + 16*(j>>2) + 8*half; // B-operand layout
        b.u[j] = *(const u32*)(wrow + kk);
      }
      acc[c] = __builtin_amdgcn_wmma_f32_16x16x32_bf16(
                 false, a.v, false, b.v, (short)0, acc[c], false, false);
    }
  }
  #pragma unroll
  for (int c=0;c<4;c++){
    int n = nt*64 + c*16 + l15;
    if (n >= N) continue;
    float bv = bias ? bias[n] : 0.0f;
    #pragma unroll
    for (int r=0;r<8;r++){
      int m = mt*16 + r + 8*half;
      float v = acc[c][r] + bv;
      if (EPI == EPI_F32){
        outF[(size_t)m*ldF + n] = v;
      } else if (EPI == EPI_SIGPOS){
        v = sigf(v);
        int seg = (m % SS) / PATCH;
        v += aux[seg*EE + n];
        outF[(size_t)m*ldF + n] = v;
        outB[(size_t)m*ldB + n] = (__bf16)v;
      } else if (EPI == EPI_RELU_BF16){
        v = v > 0.f ? v : 0.f;
        outB[(size_t)m*ldB + n] = (__bf16)v;
      } else if (EPI == EPI_RESID){
        float x = outF[(size_t)m*ldF + n] + v;
        outF[(size_t)m*ldF + n] = x;
        outB[(size_t)m*ldB + n] = (__bf16)x;
      } else if (EPI == EPI_QKV){
        int b = m / SS, sI = m % SS;
        int h = n >> 6, d = n & 63;
        outB[ (((size_t)(b*NH_ + h))*SP + sI)*64 + d ] = (__bf16)v;
      } else if (EPI == EPI_VT){
        int b = m / SS, sI = m % SS;
        int h = n >> 6, d = n & 63;
        outB[ (((size_t)(b*NH_ + h))*64 + d)*SP + sI ] = (__bf16)v;  // V transposed
      } else if (EPI == EPI_TANH){
        outF[(size_t)m*ldF + n] = tanhf(v);
      }
    }
  }
}

// ---------------- fused flash attention: one wave per (b,h, 16-row q tile) ----------------
__global__ __launch_bounds__(128) void attn_kernel(
    const __bf16* __restrict__ qb, const __bf16* __restrict__ kb,
    const __bf16* __restrict__ vt, __bf16* __restrict__ ob)
{
  __shared__ __attribute__((aligned(16))) __bf16 plds[4][16*32];
  int wv = threadIdx.x >> 5;
  int gw = blockIdx.x*4 + wv;
  int bh = gw / QT, qt = gw % QT;
  if (bh >= BB*NH_) return;
  int lane = threadIdx.x & 31, l15 = lane & 15, half = lane >> 4;
  const __bf16* qbase = qb + (size_t)bh*SP*64;
  const __bf16* kbase = kb + (size_t)bh*SP*64;
  const __bf16* vbase = vt + (size_t)bh*64*SP;

  Frag qa[2];
  {
    int qrow = qt*16 + l15;
    #pragma unroll
    for (int s2=0;s2<2;s2++)
      #pragma unroll
      for (int j=0;j<8;j++){
        int kk = s2*32 + 2*(j&3) + 16*(j>>2) + 8*half;
        qa[s2].u[j] = *(const u32*)(qbase + (size_t)qrow*64 + kk);
      }
  }
  v8f acc[4];
  #pragma unroll
  for (int c=0;c<4;c++){ v8f z = {0.f,0.f,0.f,0.f,0.f,0.f,0.f,0.f}; acc[c]=z; }
  float rmax[8], rsum[8];
  #pragma unroll
  for (int r=0;r<8;r++){ rmax[r] = -3.0e38f; rsum[r] = 0.f; }

  for (int kc=0; kc<SP; kc+=32){
    v8f cs[2];
    #pragma unroll
    for (int t2=0;t2<2;t2++){
      int krow = kc + t2*16 + l15;
      Frag k0f, k1f;
      #pragma unroll
      for (int j=0;j<8;j++){
        int kk = 2*(j&3) + 16*(j>>2) + 8*half;
        k0f.u[j] = *(const u32*)(kbase + (size_t)krow*64 + kk);
        k1f.u[j] = *(const u32*)(kbase + (size_t)krow*64 + 32 + kk);
      }
      v8f z = {0.f,0.f,0.f,0.f,0.f,0.f,0.f,0.f};
      z = __builtin_amdgcn_wmma_f32_16x16x32_bf16(false, qa[0].v, false, k0f.v, (short)0, z, false, false);
      z = __builtin_amdgcn_wmma_f32_16x16x32_bf16(false, qa[1].v, false, k1f.v, (short)0, z, false, false);
      bool msk = (krow >= SS);
      #pragma unroll
      for (int r=0;r<8;r++) cs[t2][r] = msk ? -1.0e30f : z[r]*0.125f; // 1/sqrt(64)
    }
    // online softmax: rows r+8*half live across a 16-lane group
    #pragma unroll
    for (int r=0;r<8;r++){
      float x = fmaxf(cs[0][r], cs[1][r]);
      #pragma unroll
      for (int o=1;o<16;o<<=1) x = fmaxf(x, __shfl_xor(x, o, 16));
      float mnew = fmaxf(rmax[r], x);
      float alpha = __expf(rmax[r]-mnew);
      rmax[r] = mnew;
      rsum[r] *= alpha;
      #pragma unroll
      for (int c=0;c<4;c++) acc[c][r] *= alpha;
      float p0 = __expf(cs[0][r]-mnew);
      float p1 = __expf(cs[1][r]-mnew);
      cs[0][r]=p0; cs[1][r]=p1;
      float su = p0+p1;
      #pragma unroll
      for (int o=1;o<16;o<<=1) su += __shfl_xor(su, o, 16);
      rsum[r] += su;
    }
    // P: C-layout -> LDS -> A-layout (wave-private tile, dscnt wait only)
    __bf16* pl = plds[wv];
    #pragma unroll
    for (int t2=0;t2<2;t2++)
      #pragma unroll
      for (int r=0;r<8;r++)
        pl[(r + 8*half)*32 + l15 + 16*t2] = (__bf16)cs[t2][r];
    asm volatile("s_wait_dscnt 0" ::: "memory");
    Frag pa;
    #pragma unroll
    for (int j=0;j<8;j++){
      int kk = 2*(j&3) + 16*(j>>2) + 8*half;
      pa.u[j] = *(const u32*)(pl + l15*32 + kk);
    }
    #pragma unroll
    for (int c=0;c<4;c++){
      Frag vf;
      const __bf16* vr = vbase + (size_t)(c*16 + l15)*SP + kc;
      #pragma unroll
      for (int j=0;j<8;j++){
        int kk = 2*(j&3) + 16*(j>>2) + 8*half;
        vf.u[j] = *(const u32*)(vr + kk);
      }
      acc[c] = __builtin_amdgcn_wmma_f32_16x16x32_bf16(false, pa.v, false, vf.v, (short)0, acc[c], false, false);
    }
  }
  int b = bh >> 3, h = bh & 7;
  #pragma unroll
  for (int c=0;c<4;c++)
    #pragma unroll
    for (int r=0;r<8;r++){
      int q = qt*16 + r + 8*half;
      if (q >= SS) continue;
      float o = acc[c][r] / rsum[r];
      ob[ ((size_t)(b*SS + q))*EE + h*64 + c*16 + l15 ] = (__bf16)o;
    }
}

// ---------------- GRU gate combine ----------------
__global__ void gru_combine(const float* __restrict__ gi, const float* __restrict__ gh,
                            const float* __restrict__ h, float* __restrict__ hF,
                            __bf16* __restrict__ hB, int ldb)
{
  int b = blockIdx.x, e = threadIdx.x;
  const float* gib = gi + (size_t)b*1536;
  const float* ghb = gh + (size_t)b*1536;
  float r = sigf(gib[e]       + ghb[e]);
  float z = sigf(gib[e+512]   + ghb[e+512]);
  float n = tanhf(gib[e+1024] + r*ghb[e+1024]);
  float hn = (1.f-z)*n + z*h[(size_t)b*EE + e];
  if (hF) hF[(size_t)b*EE + e] = hn;
  hB[(size_t)b*ldb + e] = (__bf16)hn;
}
__global__ void prep_concat(const float* __restrict__ h0, const float* __restrict__ h1,
                            __bf16* __restrict__ cat){
  int b = blockIdx.x, e = threadIdx.x;
  cat[(size_t)b*1024 + e]       = (__bf16)h0[(size_t)b*EE + e];
  cat[(size_t)b*1024 + 512 + e] = (__bf16)h1[(size_t)b*EE + e];
}

// ---------------- NLL over V=30000 ----------------
__global__ __launch_bounds__(256) void nll_kernel(const float* __restrict__ lg,
    const int* __restrict__ des, float* __restrict__ out)
{
  __shared__ float sh[8];
  int row = blockIdx.x;              // t*BB + b
  int t = row / BB, b = row % BB;
  int tgt = des[b*(TD+1) + t + 1];
  const float* lr = lg + (size_t)row*VV;
  float mx = -3.0e38f;
  for (int i=threadIdx.x;i<VV;i+=256) mx = fmaxf(mx, lr[i]);
  for (int o=16;o;o>>=1) mx = fmaxf(mx, __shfl_xor(mx,o,32));
  if ((threadIdx.x&31)==0) sh[threadIdx.x>>5]=mx;
  __syncthreads();
  if (threadIdx.x==0){ float m2=sh[0]; for(int i=1;i<8;i++) m2=fmaxf(m2,sh[i]); sh[0]=m2; }
  __syncthreads();
  mx = sh[0];
  __syncthreads();
  float s=0.f;
  for (int i=threadIdx.x;i<VV;i+=256) s += __expf(lr[i]-mx);
  for (int o=16;o;o>>=1) s += __shfl_xor(s,o,32);
  if ((threadIdx.x&31)==0) sh[threadIdx.x>>5]=s;
  __syncthreads();
  if (threadIdx.x==0){
    float tot=0.f; for(int i=0;i<8;i++) tot+=sh[i];
    out[row] = (tgt==0) ? 0.0f : (mx + logf(tot) - lr[tgt]);
  }
}

// ---------------- host-side GEMM dispatch ----------------
static void launch_gemm(int epi, const __bf16* A,int lda,const __bf16* W,const float* bias,
                        int M,int N,int K, float* oF,int ldF, __bf16* oB,int ldB,
                        const float* aux, hipStream_t s)
{
  int ntil = (N+63)/64;
  int waves = (M/16)*ntil;
  dim3 g((waves+7)/8), bl(256);
  switch(epi){
    case EPI_F32:       gemm_bf16<EPI_F32>      <<<g,bl,0,s>>>(A,lda,W,bias,M,N,K,oF,ldF,oB,ldB,aux); break;
    case EPI_SIGPOS:    gemm_bf16<EPI_SIGPOS>   <<<g,bl,0,s>>>(A,lda,W,bias,M,N,K,oF,ldF,oB,ldB,aux); break;
    case EPI_RELU_BF16: gemm_bf16<EPI_RELU_BF16><<<g,bl,0,s>>>(A,lda,W,bias,M,N,K,oF,ldF,oB,ldB,aux); break;
    case EPI_RESID:     gemm_bf16<EPI_RESID>    <<<g,bl,0,s>>>(A,lda,W,bias,M,N,K,oF,ldF,oB,ldB,aux); break;
    case EPI_QKV:       gemm_bf16<EPI_QKV>      <<<g,bl,0,s>>>(A,lda,W,bias,M,N,K,oF,ldF,oB,ldB,aux); break;
    case EPI_VT:        gemm_bf16<EPI_VT>       <<<g,bl,0,s>>>(A,lda,W,bias,M,N,K,oF,ldF,oB,ldB,aux); break;
    case EPI_TANH:      gemm_bf16<EPI_TANH>     <<<g,bl,0,s>>>(A,lda,W,bias,M,N,K,oF,ldF,oB,ldB,aux); break;
  }
}

extern "C" void kernel_launch(void* const* d_in, const int* in_sizes, int n_in,
                              void* d_out, int out_size, void* d_ws, size_t ws_size,
                              hipStream_t stream)
{
  (void)in_sizes; (void)n_in; (void)out_size; (void)ws_size;
  const float* Img1 = (const float*)d_in[0];
  const float* Img2 = (const float*)d_in[1];
  const int*   Des  = (const int*)  d_in[2];
  const float* word_emb = (const float*)d_in[3];
  const float* pos_emb  = (const float*)d_in[4];
  const float* proj_w = (const float*)d_in[5];
  const float* proj_b = (const float*)d_in[6];
  const float* enc_a  = (const float*)d_in[7];
  const float* enc_b  = (const float*)d_in[8];
  const float* attn_w = (const float*)d_in[9];
  const float* attn_b = (const float*)d_in[10];
  const float* ffn_w1 = (const float*)d_in[11];
  const float* ffn_b1 = (const float*)d_in[12];
  const float* ffn_w2 = (const float*)d_in[13];
  const float* ffn_b2 = (const float*)d_in[14];
  const float* sn_a   = (const float*)d_in[15];
  const float* sn_b   = (const float*)d_in[16];
  const float* g0wih  = (const float*)d_in[17];
  const float* g0whh  = (const float*)d_in[18];
  const float* g1wih  = (const float*)d_in[19];
  const float* g1whh  = (const float*)d_in[20];
  const float* dlw    = (const float*)d_in[21];
  const float* dna    = (const float*)d_in[22];
  const float* dnb    = (const float*)d_in[23];
  const float* outw   = (const float*)d_in[24];
  const float* outb   = (const float*)d_in[25];

  char* base = (char*)d_ws; size_t off = 0;
  auto alloc = [&](size_t bytes)->void*{ void* r = base+off; off = (off+bytes+255)&~(size_t)255; return r; };

  __bf16* Abf  = (__bf16*)alloc((size_t)MM*CH*2);
  float*  Xf   = (float*) alloc((size_t)MM*EE*4);
  __bf16* Xbf  = (__bf16*)alloc((size_t)MM*EE*2);
  __bf16* Ybf  = (__bf16*)alloc((size_t)MM*EE*2);
  __bf16* Hbf  = (__bf16*)alloc((size_t)MM*EE*2);
  __bf16* Obf  = (__bf16*)alloc((size_t)MM*EE*2);
  __bf16* qbuf = (__bf16*)alloc((size_t)BB*NH_*SP*64*2);
  __bf16* kbuf = (__bf16*)alloc((size_t)BB*NH_*SP*64*2);
  __bf16* vtb  = (__bf16*)alloc((size_t)BB*NH_*64*SP*2);
  __bf16* w_proj=(__bf16*)alloc((size_t)EE*CH*2);
  __bf16* w_attn=(__bf16*)alloc((size_t)3*4*EE*EE*2);
  __bf16* w_f1 = (__bf16*)alloc((size_t)3*EE*EE*2);
  __bf16* w_f2 = (__bf16*)alloc((size_t)3*EE*EE*2);
  __bf16* w_g0i= (__bf16*)alloc((size_t)1536*EE*2);
  __bf16* w_g0h= (__bf16*)alloc((size_t)1536*EE*2);
  __bf16* w_g1i= (__bf16*)alloc((size_t)1536*EE*2);
  __bf16* w_g1h= (__bf16*)alloc((size_t)1536*EE*2);
  __bf16* w_dl = (__bf16*)alloc((size_t)EE*2*EE*2);
  __bf16* w_out= (__bf16*)alloc((size_t)VV*EE*2);
  __bf16* xsb  = (__bf16*)alloc((size_t)TD*BB*EE*2);
  float*  Gf   = (float*) alloc((size_t)BB*EE*4);
  float*  h1f  = (float*) alloc((size_t)BB*EE*4);
  __bf16* cat  = (__bf16*)alloc((size_t)BB*1024*2);
  float*  gi0  = (float*) alloc((size_t)BB*1536*4);
  float*  gh0  = (float*) alloc((size_t)BB*1536*4);
  float*  gi1  = (float*) alloc((size_t)BB*1536*4);
  float*  gh1  = (float*) alloc((size_t)BB*1536*4);
  __bf16* g1b  = (__bf16*)alloc((size_t)BB*EE*2);
  float*  outs = (float*) alloc((size_t)TD*BB*EE*4);
  __bf16* outsb= (__bf16*)alloc((size_t)TD*BB*EE*2);
  float*  logits=(float*) alloc((size_t)TD*BB*VV*4);

  auto conv = [&](const float* s, __bf16* d, size_t n){
    cvt_f32_bf16<<<(unsigned)((n+255)/256),256,0,stream>>>(s,d,n);
  };
  conv(proj_w, w_proj, (size_t)EE*CH);
  conv(attn_w, w_attn, (size_t)3*4*EE*EE);
  conv(ffn_w1, w_f1,   (size_t)3*EE*EE);
  conv(ffn_w2, w_f2,   (size_t)3*EE*EE);
  conv(g0wih,  w_g0i,  (size_t)1536*EE);
  conv(g0whh,  w_g0h,  (size_t)1536*EE);
  conv(g1wih,  w_g1i,  (size_t)1536*EE);
  conv(g1whh,  w_g1h,  (size_t)1536*EE);
  conv(dlw,    w_dl,   (size_t)EE*2*EE);
  conv(outw,   w_out,  (size_t)VV*EE);

  // ---- encoder ----
  pack_img<<<(unsigned)(((size_t)MM*CH+255)/256),256,0,stream>>>(Img1, Img2, Abf);
  launch_gemm(EPI_SIGPOS, Abf,CH, w_proj, proj_b, MM,EE,CH, Xf,EE, Xbf,EE, pos_emb, stream);
  ln_kernel<<<MM,256,0,stream>>>(Xf, enc_a, enc_b, Xf, Xbf);   // enc LN replaces x

  int attnBlocks = (BB*NH_*QT + 3)/4;
  for (int i=0;i<3;i++){
    const __bf16* wq = w_attn + (size_t)(i*4+0)*EE*EE;
    const __bf16* wk = w_attn + (size_t)(i*4+1)*EE*EE;
    const __bf16* wv = w_attn + (size_t)(i*4+2)*EE*EE;
    const __bf16* wo = w_attn + (size_t)(i*4+3)*EE*EE;
    const float* bq = attn_b + (size_t)(i*4+0)*EE;
    const float* bk = attn_b + (size_t)(i*4+1)*EE;
    const float* bv = attn_b + (size_t)(i*4+2)*EE;
    const float* bo = attn_b + (size_t)(i*4+3)*EE;
    ln_kernel<<<MM,256,0,stream>>>(Xf, sn_a+(size_t)(i*2+0)*EE, sn_b+(size_t)(i*2+0)*EE, nullptr, Ybf);
    launch_gemm(EPI_QKV, Ybf,EE, wq, bq, MM,EE,EE, nullptr,0, qbuf,0, nullptr, stream);
    launch_gemm(EPI_QKV, Xbf,EE, wk, bk, MM,EE,EE, nullptr,0, kbuf,0, nullptr, stream);
    launch_gemm(EPI_VT , Xbf,EE, wv, bv, MM,EE,EE, nullptr,0, vtb ,0, nullptr, stream);
    attn_kernel<<<attnBlocks,128,0,stream>>>(qbuf, kbuf, vtb, Obf);
    launch_gemm(EPI_RESID, Obf,EE, wo, bo, MM,EE,EE, Xf,EE, Xbf,EE, nullptr, stream);
    ln_kernel<<<MM,256,0,stream>>>(Xf, sn_a+(size_t)(i*2+1)*EE, sn_b+(size_t)(i*2+1)*EE, nullptr, Ybf);
    launch_gemm(EPI_RELU_BF16, Ybf,EE, w_f1+(size_t)i*EE*EE, ffn_b1+(size_t)i*EE, MM,EE,EE, nullptr,0, Hbf,EE, nullptr, stream);
    launch_gemm(EPI_RESID,     Hbf,EE, w_f2+(size_t)i*EE*EE, ffn_b2+(size_t)i*EE, MM,EE,EE, Xf,EE, Xbf,EE, nullptr, stream);
  }

  // ---- decoder ----
  mean_rows<<<BB,512,0,stream>>>(Xf, Gf);
  copy_f32<<<(BB*EE+255)/256,256,0,stream>>>(Gf, h1f, (size_t)BB*EE);
  gather_emb<<<TD*BB,512,0,stream>>>(word_emb, Des, xsb);

  for (int t=0;t<TD;t++){
    const float* h0src = (t==0) ? Gf : (outs + (size_t)(t-1)*BB*EE);
    prep_concat<<<BB,512,0,stream>>>(h0src, h1f, cat);
    launch_gemm(EPI_F32, xsb+(size_t)t*BB*EE,EE, w_g0i, nullptr, BB,1536,EE, gi0,1536, nullptr,0, nullptr, stream);
    launch_gemm(EPI_F32, cat,1024,            w_g0h, nullptr, BB,1536,EE, gh0,1536, nullptr,0, nullptr, stream);
    gru_combine<<<BB,512,0,stream>>>(gi0, gh0, h0src, nullptr, g1b, EE);
    launch_gemm(EPI_F32, g1b,EE,              w_g1i, nullptr, BB,1536,EE, gi1,1536, nullptr,0, nullptr, stream);
    launch_gemm(EPI_F32, cat+512,1024,        w_g1h, nullptr, BB,1536,EE, gh1,1536, nullptr,0, nullptr, stream);
    gru_combine<<<BB,512,0,stream>>>(gi1, gh1, h1f, h1f, cat+512, 1024);  // g2 -> h1, cat[:,512:]
    launch_gemm(EPI_TANH, cat,1024, w_dl, nullptr, BB,EE,2*EE, outs+(size_t)t*BB*EE,EE, nullptr,0, nullptr, stream);
  }

  // ---- output head ----
  ln_kernel<<<TD*BB,256,0,stream>>>(outs, dna, dnb, nullptr, outsb);
  launch_gemm(EPI_F32, outsb,EE, w_out, outb, TD*BB, VV, EE, logits, VV, nullptr,0, nullptr, stream);
  nll_kernel<<<TD*BB,256,0,stream>>>(logits, Des, (float*)d_out);
}
